// SelfAttention_40046275068496
// MI455X (gfx1250) — matrix operand order repaired and checked by reading, not compiled, not run on hardware
//
#include <hip/hip_runtime.h>
#include <hip/hip_bf16.h>

// Problem constants (match reference)
#define B_   4
#define N_   2048
#define DIM_ 1024
#define H_   16
#define DH_  64
#define E3_  (3 * DIM_)   // 3072 qkv outputs
#define M_   (B_ * N_)    // 8192 total rows

typedef __attribute__((ext_vector_type(16))) __bf16 v16bf;
typedef __attribute__((ext_vector_type(8)))  __bf16 v8bf;
typedef __attribute__((ext_vector_type(8)))  float  v8f;
typedef __attribute__((ext_vector_type(4)))  float  v4f;

// ---------------------------------------------------------------------------
// WMMA helper: D = A(16x32 bf16) * B(32x16 bf16) + C(16x16 f32)
// ---------------------------------------------------------------------------
__device__ __forceinline__ v8f wmma_bf16(v16bf a, v16bf b, v8f c) {
  return __builtin_amdgcn_wmma_f32_16x16x32_bf16(
      /*neg_a=*/false, a, /*neg_b=*/false, b,
      /*c_mod=*/(short)0, c, /*reuse_a=*/false, /*reuse_b=*/false);
}

// A fragment (16x32, row-major source, leading dim ld in elements).
// ISA layout: lanes 0-15 hold row M=lane, elems = K{0..7,16..23};
//             lanes 16-31 hold row M=lane-16, elems = K{8..15,24..31}.
__device__ __forceinline__ v16bf loadA(const __bf16* p, int ld, int lane) {
  const int m  = lane & 15;
  const int kh = (lane >> 4) & 1;
  const __bf16* q = p + (size_t)m * ld + kh * 8;
  v8bf lo = *(const v8bf*)(q);        // K = kh*8 + 0..7
  v8bf hi = *(const v8bf*)(q + 16);   // K = 16 + kh*8 + 0..7
  return __builtin_shufflevector(lo, hi, 0, 1, 2, 3, 4, 5, 6, 7,
                                 8, 9, 10, 11, 12, 13, 14, 15);
}

// B fragment (32x16): element (k, n) lives at p + n*ld + k  (per-column-
// contiguous K). ISA layout: lanes 0-15 col N=lane hold K{0..7,16..23}; lanes
// 16-31 col N=lane-16 hold K{8..15,24..31}. Two 16-byte loads per lane.
__device__ __forceinline__ v16bf loadB(const __bf16* p, int ld, int lane) {
  const int n  = lane & 15;
  const int kh = (lane >> 4) & 1;
  const __bf16* q = p + (size_t)n * ld + kh * 8;
  v8bf lo = *(const v8bf*)(q);
  v8bf hi = *(const v8bf*)(q + 16);
  return __builtin_shufflevector(lo, hi, 0, 1, 2, 3, 4, 5, 6, 7,
                                 8, 9, 10, 11, 12, 13, 14, 15);
}

// Fragment set for one 32(M) x 64(N) x 32(K) GEMM step.
struct GemmFrags {
  v16bf a[2];
  v16bf b[4];
};

__device__ __forceinline__ void load_frags(GemmFrags& f,
                                           const __bf16* __restrict__ X,
                                           const __bf16* __restrict__ W,
                                           int row0, int col0, int k, int lane) {
#pragma unroll
  for (int mi = 0; mi < 2; ++mi)
    f.a[mi] = loadA(X + (size_t)(row0 + mi * 16) * DIM_ + k, DIM_, lane);
#pragma unroll
  for (int t = 0; t < 4; ++t)
    f.b[t] = loadB(W + (size_t)(col0 + t * 16) * DIM_ + k, DIM_, lane);
}

// ---------------------------------------------------------------------------
// Kernel 1: fp32 -> bf16 conversion (4 elements / thread)
// ---------------------------------------------------------------------------
__global__ __launch_bounds__(256) void cvt_f32_bf16(const float* __restrict__ in,
                                                    __bf16* __restrict__ out,
                                                    int n4) {
  int i = blockIdx.x * 256 + threadIdx.x;
  if (i < n4) {
    v4f f = *(const v4f*)(in + (size_t)i * 4);
    __bf16 b4[4] = {(__bf16)f.x, (__bf16)f.y, (__bf16)f.z, (__bf16)f.w};
    *(unsigned long long*)(out + (size_t)i * 4) = *(unsigned long long*)b4;
  }
}

// ---------------------------------------------------------------------------
// Kernel 2: QKV projection.  C[m,e] = sum_d X[m,d] * Wqkv[e,d]
// Wave = 32(M) x 64(E) tile.  k-loop unrolled by 2 with role-alternating
// fragment buffers: phase p consumes fr[p] then refills fr[p] for k+64.
// No copy chain; each phase's WMMAs wait only for its own (older) loads.
// Scatter: Q,K -> [B,H,N,DH] bf16 ; V -> [B,H,DH,N] bf16 (transposed).
// ---------------------------------------------------------------------------
__global__ __launch_bounds__(256) void qkv_gemm(const __bf16* __restrict__ X,
                                                const __bf16* __restrict__ W,
                                                __bf16* __restrict__ Q,
                                                __bf16* __restrict__ K,
                                                __bf16* __restrict__ V) {
  const int lane = threadIdx.x & 31;
  const int wave = threadIdx.x >> 5;
  const int row0 = blockIdx.y * 128 + (wave & 3) * 32;   // M tile (32 rows)
  const int col0 = blockIdx.x * 128 + (wave >> 2) * 64;  // E tile (64 cols)

  const v8f z = {0, 0, 0, 0, 0, 0, 0, 0};
  v8f acc[2][4] = {{z, z, z, z}, {z, z, z, z}};

  GemmFrags fr[2];
  load_frags(fr[0], X, W, row0, col0, 0, lane);
  load_frags(fr[1], X, W, row0, col0, 32, lane);

  for (int k = 0; k < DIM_; k += 64) {
#pragma unroll
    for (int ph = 0; ph < 2; ++ph) {
#pragma unroll
      for (int mi = 0; mi < 2; ++mi)
#pragma unroll
        for (int t = 0; t < 4; ++t)
          acc[mi][t] = wmma_bf16(fr[ph].a[mi], fr[ph].b[t], acc[mi][t]);
      const int kk = k + ph * 32;
      const int kn = (kk + 64 < DIM_) ? kk + 64 : ph * 32;  // clamped refill
      load_frags(fr[ph], X, W, row0, col0, kn, lane);
    }
  }

  const int mhalf = (lane >> 4) * 8;
  const int nc = lane & 15;
#pragma unroll
  for (int mi = 0; mi < 2; ++mi) {
#pragma unroll
    for (int t = 0; t < 4; ++t) {
      const int e     = col0 + t * 16 + nc;   // 0..3071
      const int which = e >> 10;              // 0=Q 1=K 2=V
      const int eh    = e & (DIM_ - 1);
      const int h     = eh >> 6;
      const int d     = eh & (DH_ - 1);
#pragma unroll
      for (int r = 0; r < 8; ++r) {
        const int row = row0 + mi * 16 + mhalf + r;
        const int bb  = row >> 11;            // / N_
        const int n   = row & (N_ - 1);
        const __bf16 val = (__bf16)acc[mi][t][r];
        if (which == 0)
          Q[(((size_t)bb * H_ + h) * N_ + n) * DH_ + d] = val;
        else if (which == 1)
          K[(((size_t)bb * H_ + h) * N_ + n) * DH_ + d] = val;
        else
          V[(((size_t)bb * H_ + h) * DH_ + d) * N_ + n] = val;
      }
    }
  }
}

// ---------------------------------------------------------------------------
// Kernel 3: flash attention, online softmax.
// Block = 128 threads = 4 waves; wave owns 16 query rows; KV step = 32,
// j-loop unrolled by 2 with role-alternating K-fragment buffers (no copies).
// V loads are issued at the top of each step; the K refill for step+2 flies
// under the softmax VALU section and the LDS P staging.
// ---------------------------------------------------------------------------
__global__ __launch_bounds__(128) void flash_attn(const __bf16* __restrict__ Q,
                                                  const __bf16* __restrict__ K,
                                                  const __bf16* __restrict__ V,
                                                  __bf16* __restrict__ O) {
  __shared__ __bf16 pbuf[4][16 * 32];   // per-wave P staging (C->A relayout)

  const int lane = threadIdx.x & 31;
  const int wave = threadIdx.x >> 5;
  const int h = blockIdx.y;
  const int b = blockIdx.z;
  const int q0 = blockIdx.x * 64 + wave * 16;

  const size_t qk_base = ((size_t)b * H_ + h) * N_ * DH_;  // Q,K [B,H,N,DH]
  const size_t v_base  = ((size_t)b * H_ + h) * DH_ * N_;  // V   [B,H,DH,N]

  const int mhalf = (lane >> 4) * 8;
  const int nc = lane & 15;
  const float scale = 0.125f;   // DH^-0.5

  // Q fragments (16 rows x 64 d), in registers for the whole pass.
  v16bf qa0 = loadA(Q + qk_base + (size_t)q0 * DH_ + 0,  DH_, lane);
  v16bf qa1 = loadA(Q + qk_base + (size_t)q0 * DH_ + 32, DH_, lane);

  const v8f z = {0, 0, 0, 0, 0, 0, 0, 0};
  v8f oacc[4] = {z, z, z, z};
  float mrow[8], lrow[8];
#pragma unroll
  for (int r = 0; r < 8; ++r) { mrow[r] = -1e30f; lrow[r] = 0.f; }

  __bf16* pl = pbuf[wave];

  // K-fragment double buffer: kbuf[p] holds the tile for step (j0 + p*32).
  v16bf kbuf[2][4];
#pragma unroll
  for (int ph = 0; ph < 2; ++ph) {
    const int jj = ph * 32;
    kbuf[ph][0] = loadB(K + qk_base + (size_t)jj * DH_ +  0, DH_, lane);
    kbuf[ph][1] = loadB(K + qk_base + (size_t)jj * DH_ + 32, DH_, lane);
    kbuf[ph][2] = loadB(K + qk_base + (size_t)(jj + 16) * DH_ +  0, DH_, lane);
    kbuf[ph][3] = loadB(K + qk_base + (size_t)(jj + 16) * DH_ + 32, DH_, lane);
  }

  for (int j0 = 0; j0 < N_; j0 += 64) {
#pragma unroll
    for (int ph = 0; ph < 2; ++ph) {
      const int jj = j0 + ph * 32;

      // V-tile loads first: independent, overlap score WMMAs + softmax.
      v16bf vb[4];
#pragma unroll
      for (int t = 0; t < 4; ++t)
        vb[t] = loadB(V + v_base + (size_t)(t * 16) * N_ + jj, N_, lane);

      // ---- scores S = (Q @ K^T) * scale : two 16x16 tiles ----
      v8f s0 = z, s1 = z;
      s0 = wmma_bf16(qa0, kbuf[ph][0], s0);
      s0 = wmma_bf16(qa1, kbuf[ph][1], s0);
      s1 = wmma_bf16(qa0, kbuf[ph][2], s1);
      s1 = wmma_bf16(qa1, kbuf[ph][3], s1);

      // Refill this phase's K buffer for step jj+64 (clamped); completes
      // under the softmax VALU work below.
      const int jn = (jj + 64 < N_) ? jj + 64 : ph * 32;
      kbuf[ph][0] = loadB(K + qk_base + (size_t)jn * DH_ +  0, DH_, lane);
      kbuf[ph][1] = loadB(K + qk_base + (size_t)jn * DH_ + 32, DH_, lane);
      kbuf[ph][2] = loadB(K + qk_base + (size_t)(jn + 16) * DH_ +  0, DH_, lane);
      kbuf[ph][3] = loadB(K + qk_base + (size_t)(jn + 16) * DH_ + 32, DH_, lane);

      // ---- online softmax (row = (lane>>4)*8 + r, cols across 16 lanes) ----
      float corr[8];
#pragma unroll
      for (int r = 0; r < 8; ++r) {
        float a = s0[r] * scale, c = s1[r] * scale;
        float t = fmaxf(a, c);
        t = fmaxf(t, __shfl_xor(t, 1));
        t = fmaxf(t, __shfl_xor(t, 2));
        t = fmaxf(t, __shfl_xor(t, 4));
        t = fmaxf(t, __shfl_xor(t, 8));
        const float nm = fmaxf(mrow[r], t);
        const float p0 = __expf(a - nm);
        const float p1 = __expf(c - nm);
        s0[r] = p0; s1[r] = p1;
        float su = p0 + p1;
        su += __shfl_xor(su, 1);
        su += __shfl_xor(su, 2);
        su += __shfl_xor(su, 4);
        su += __shfl_xor(su, 8);
        corr[r] = __expf(mrow[r] - nm);
        lrow[r] = lrow[r] * corr[r] + su;
        mrow[r] = nm;
      }
#pragma unroll
      for (int t = 0; t < 4; ++t)
#pragma unroll
        for (int r = 0; r < 8; ++r) oacc[t][r] *= corr[r];

      // ---- stage P (C layout) into LDS, reload as A fragment ----
#pragma unroll
      for (int r = 0; r < 8; ++r) {
        pl[(mhalf + r) * 32 + nc]      = (__bf16)s0[r];
        pl[(mhalf + r) * 32 + 16 + nc] = (__bf16)s1[r];
      }
      __syncthreads();   // order LDS writes vs cross-lane reads
      v16bf pa = loadA(pl, 32, lane);

      // ---- O += P(16x32) @ V(32x64) ; V is [DH,N] so K is contiguous ----
#pragma unroll
      for (int t = 0; t < 4; ++t)
        oacc[t] = wmma_bf16(pa, vb[t], oacc[t]);
      __syncthreads();   // protect pl before next step's writes
    }
  }

  // ---- normalize, store O as bf16 [B, N, H*DH] for the out-projection ----
#pragma unroll
  for (int r = 0; r < 8; ++r) lrow[r] = __frcp_rn(lrow[r]);
#pragma unroll
  for (int t = 0; t < 4; ++t) {
#pragma unroll
    for (int r = 0; r < 8; ++r) {
      const int n = q0 + mhalf + r;
      const float ov = oacc[t][r] * lrow[r];
      O[((size_t)b * N_ + n) * DIM_ + h * DH_ + t * 16 + nc] = (__bf16)ov;
    }
  }
}

// ---------------------------------------------------------------------------
// Kernel 4: out projection  Y[m,n] = sum_k O[m,k]*Wout[n,k] + bias[n]  (fp32)
// Same 32x64-per-wave phase-alternating structure as qkv_gemm.
// ---------------------------------------------------------------------------
__global__ __launch_bounds__(256) void out_gemm(const __bf16* __restrict__ A,
                                                const __bf16* __restrict__ W,
                                                const float* __restrict__ bias,
                                                float* __restrict__ Y) {
  const int lane = threadIdx.x & 31;
  const int wave = threadIdx.x >> 5;
  const int row0 = blockIdx.y * 128 + (wave & 3) * 32;
  const int col0 = blockIdx.x * 128 + (wave >> 2) * 64;

  const v8f z = {0, 0, 0, 0, 0, 0, 0, 0};
  v8f acc[2][4] = {{z, z, z, z}, {z, z, z, z}};

  GemmFrags fr[2];
  load_frags(fr[0], A, W, row0, col0, 0, lane);
  load_frags(fr[1], A, W, row0, col0, 32, lane);

  for (int k = 0; k < DIM_; k += 64) {
#pragma unroll
    for (int ph = 0; ph < 2; ++ph) {
#pragma unroll
      for (int mi = 0; mi < 2; ++mi)
#pragma unroll
        for (int t = 0; t < 4; ++t)
          acc[mi][t] = wmma_bf16(fr[ph].a[mi], fr[ph].b[t], acc[mi][t]);
      const int kk = k + ph * 32;
      const int kn = (kk + 64 < DIM_) ? kk + 64 : ph * 32;
      load_frags(fr[ph], A, W, row0, col0, kn, lane);
    }
  }

  const int mhalf = (lane >> 4) * 8;
  const int nc = lane & 15;
#pragma unroll
  for (int mi = 0; mi < 2; ++mi) {
#pragma unroll
    for (int t = 0; t < 4; ++t) {
      const int col = col0 + t * 16 + nc;
#pragma unroll
      for (int r = 0; r < 8; ++r) {
        const int row = row0 + mi * 16 + mhalf + r;
        Y[(size_t)row * DIM_ + col] = acc[mi][t][r] + bias[col];
      }
    }
  }
}

// ---------------------------------------------------------------------------
// Host launcher
// ---------------------------------------------------------------------------
extern "C" void kernel_launch(void* const* d_in, const int* in_sizes, int n_in,
                              void* d_out, int out_size, void* d_ws, size_t ws_size,
                              hipStream_t stream) {
  (void)in_sizes; (void)n_in; (void)out_size; (void)ws_size;
  const float* x     = (const float*)d_in[0];
  const float* w_qkv = (const float*)d_in[1];
  const float* w_out = (const float*)d_in[2];
  const float* b_out = (const float*)d_in[3];
  float* y = (float*)d_out;

  // Workspace carve-out (all bf16), 256B aligned. Total ~90 MB.
  char* ws = (char*)d_ws;
  size_t off = 0;
  auto carve = [&](size_t bytes) -> void* {
    void* p = ws + off;
    off += (bytes + 255) & ~(size_t)255;
    return p;
  };
  __bf16* xb  = (__bf16*)carve((size_t)M_  * DIM_ * 2);
  __bf16* wqb = (__bf16*)carve((size_t)E3_ * DIM_ * 2);
  __bf16* wob = (__bf16*)carve((size_t)DIM_ * DIM_ * 2);
  __bf16* Qb  = (__bf16*)carve((size_t)M_ * DIM_ * 2);
  __bf16* Kb  = (__bf16*)carve((size_t)M_ * DIM_ * 2);
  __bf16* Vb  = (__bf16*)carve((size_t)M_ * DIM_ * 2);
  __bf16* Ob  = (__bf16*)carve((size_t)M_ * DIM_ * 2);

  const int nx = M_ * DIM_ / 4, nwq = E3_ * DIM_ / 4, nwo = DIM_ * DIM_ / 4;
  cvt_f32_bf16<<<(nx  + 255) / 256, 256, 0, stream>>>(x,     xb,  nx);
  cvt_f32_bf16<<<(nwq + 255) / 256, 256, 0, stream>>>(w_qkv, wqb, nwq);
  cvt_f32_bf16<<<(nwo + 255) / 256, 256, 0, stream>>>(w_out, wob, nwo);

  qkv_gemm<<<dim3(E3_ / 128, M_ / 128), 256, 0, stream>>>(xb, wqb, Qb, Kb, Vb);
  flash_attn<<<dim3(N_ / 64, H_, B_), 128, 0, stream>>>(Qb, Kb, Vb, Ob);
  out_gemm<<<dim3(DIM_ / 128, M_ / 128), 256, 0, stream>>>(Ob, wob, b_out, y);
}
